// TaxoRec_80839874445975
// MI455X (gfx1250) — compile-verified, matrix operand order checked by reading, not run on hardware
//
#include <hip/hip_runtime.h>
#include <math.h>

typedef __attribute__((ext_vector_type(16))) __bf16 v16bf;
typedef __attribute__((ext_vector_type(8)))  float  v8f;

#define EPSF 1e-15f

// ---------------- utility kernels ----------------

__global__ void k_zero(float* __restrict__ p, long long n) {
  long long i = (long long)blockIdx.x * blockDim.x + threadIdx.x;
  if (i < n) p[i] = 0.f;
}

__global__ void k_accum(float* __restrict__ acc, const float* __restrict__ add, long long n) {
  long long i = (long long)blockIdx.x * blockDim.x + threadIdx.x;
  if (i < n) acc[i] += add[i];
}

// ---------------- fused projx + logmap0 ----------------
__global__ void k_projx_logmap(const float* __restrict__ src, float* __restrict__ dst, int nrows) {
  int r = blockIdx.x * blockDim.x + threadIdx.x;
  if (r >= nrows) return;
  const float* x = src + (long long)r * 64;
  float s = 0.f;
#pragma unroll
  for (int j = 1; j < 64; ++j) { float v = x[j]; s += v * v; }
  float x0    = sqrtf(1.f + s);
  float alpha = fmaxf(x0, 1.f + 1e-7f);
  float rn    = sqrtf(fmaxf(s, EPSF));
  float sc    = acoshf(alpha) / rn;
  float* o = dst + (long long)r * 64;
  o[0] = 0.f;
#pragma unroll
  for (int j = 1; j < 64; ++j) o[j] = sc * x[j];
}

// ---------------- edge scatter: nxt[dst] += w * cur[src] ----------------
// wave32 = 32 contiguous columns of one edge -> coalesced loads + L2-resident atomics.
__global__ void k_scatter(const float* __restrict__ cur, float* __restrict__ nxt,
                          const int* __restrict__ esrc, const int* __restrict__ edst,
                          const float* __restrict__ ew, long long E) {
  long long t = (long long)blockIdx.x * blockDim.x + threadIdx.x;
  long long e = t >> 6;
  if (e >= E) return;
  int c = (int)(t & 63);
  int s = esrc[e], d = edst[e];
  float w = ew[e];
  atomicAdd(&nxt[(long long)d * 64 + c], w * cur[(long long)s * 64 + c]);
}

// ---------------- expmap0 ----------------
__global__ void k_expmap(const float* __restrict__ acc, float* __restrict__ h, int nrows) {
  int r = blockIdx.x * blockDim.x + threadIdx.x;
  if (r >= nrows) return;
  const float* u = acc + (long long)r * 64;
  float s = 0.f;
#pragma unroll
  for (int j = 1; j < 64; ++j) { float v = u[j]; s += v * v; }
  float nrm = sqrtf(fmaxf(s, EPSF));
  float sc  = sinhf(nrm) / nrm;
  float* o = h + (long long)r * 64;
  o[0] = coshf(nrm);
#pragma unroll
  for (int j = 1; j < 64; ++j) o[j] = sc * u[j];
}

// ---------------- tag prep: M[t] = [gamma*k (63) | gamma] ----------------
__global__ void k_tagprep(const float* __restrict__ T, float* __restrict__ M, int tags) {
  int t = blockIdx.x * blockDim.x + threadIdx.x;
  if (t >= tags) return;
  const float* x = T + (long long)t * 64;
  float s = 0.f;
#pragma unroll
  for (int j = 1; j < 64; ++j) { float v = x[j]; s += v * v; }
  float x0  = sqrtf(1.f + s);
  float inv = 1.f / (x0 + 1.f);
  float np  = s * inv * inv;
  float kf  = 2.f * inv / (1.f + np);
  float nk  = s * kf * kf;
  float gamma = 1.f / sqrtf(fmaxf(1.f - nk, EPSF));
  float* o = M + (long long)t * 64;
#pragma unroll
  for (int j = 1; j < 64; ++j) o[j - 1] = gamma * kf * x[j];
  o[63] = gamma;
}

// ---------------- WMMA GEMM: C(NI x 64) = sps(NI x K) @ M(K x 64) ----------------
// 256 threads = 8 waves; block tile 128x64; per-wave 16 rows x 64 cols (4 v8f accs).
//
// B tile (32 k x 64 n) is stored in LDS PRE-SWIZZLED into the WMMA B-fragment
// layout: Bp[n][32] bf16, where 16-element slot pairs are permuted so that the
// fragment a lane needs (k = {kh*8+0..7(step2 pairs)} U {16+kh*8+...}) occupies
// 32 contiguous, 16B-aligned bytes -> two ds_load_b128 per fragment.
// slot s -> k-pair p = (s&3) + ((s>>2)&1)*8 + (s>>3)*4   (k = 2p, 2p+1)
//   kh=0 reads slots 0..7  -> k = 0,2,4,6,16,18,20,22 (+pair partner)
//   kh=1 reads slots 8..15 -> k = 8,10,12,14,24,26,28,30
__global__ void k_gemm_tags(const float* __restrict__ A, const float* __restrict__ Bm,
                            float* __restrict__ C, int K) {
  __shared__ __bf16 Bp[64][32];     // [n][2*slot] = bf16 pairs, 4 KB
  const int tid  = threadIdx.x;
  const int wave = tid >> 5;
  const int lane = tid & 31;
  const int mh   = lane & 15;       // M (A/C) or N (B) within tile
  const int kh   = lane >> 4;       // K-half selector
  const int mBase = blockIdx.x * 128 + wave * 16;

  union FragU { uint4 u[2]; v16bf v; };

  v8f acc0 = {}, acc1 = {}, acc2 = {}, acc3 = {};

  for (int k0 = 0; k0 < K; k0 += 32) {
    // cooperative B tile load: coalesced in n, swizzled store into fragment layout
#pragma unroll
    for (int t = 0; t < 4; ++t) {
      int e = tid + t * 256;                      // 0..1023 pair entries
      int s = e >> 6;                             // slot 0..15
      int n = e & 63;
      int p = (s & 3) + ((s >> 2) & 1) * 8 + (s >> 3) * 4;
      int k = k0 + 2 * p;
      Bp[n][2 * s]     = (__bf16)Bm[(long long)k * 64 + n];
      Bp[n][2 * s + 1] = (__bf16)Bm[(long long)(k + 1) * 64 + n];
    }
    __syncthreads();

    // A fragment: two contiguous 32B runs per lane -> 4x global_load_b128 + pk cvt
    const float* arow = A + (long long)(mBase + mh) * K + k0 + kh * 8;
    float4 a0 = ((const float4*)arow)[0];
    float4 a1 = ((const float4*)arow)[1];
    float4 a2 = ((const float4*)(arow + 16))[0];
    float4 a3 = ((const float4*)(arow + 16))[1];
    v16bf af;
    af[0]  = (__bf16)a0.x; af[1]  = (__bf16)a0.y; af[2]  = (__bf16)a0.z; af[3]  = (__bf16)a0.w;
    af[4]  = (__bf16)a1.x; af[5]  = (__bf16)a1.y; af[6]  = (__bf16)a1.z; af[7]  = (__bf16)a1.w;
    af[8]  = (__bf16)a2.x; af[9]  = (__bf16)a2.y; af[10] = (__bf16)a2.z; af[11] = (__bf16)a2.w;
    af[12] = (__bf16)a3.x; af[13] = (__bf16)a3.y; af[14] = (__bf16)a3.z; af[15] = (__bf16)a3.w;

    // B fragments: 2x ds_load_b128 each for the 4 column tiles
    FragU fb0, fb1, fb2, fb3;
    {
      const uint4* q0 = (const uint4*)(&Bp[ 0 + mh][kh * 16]);
      const uint4* q1 = (const uint4*)(&Bp[16 + mh][kh * 16]);
      const uint4* q2 = (const uint4*)(&Bp[32 + mh][kh * 16]);
      const uint4* q3 = (const uint4*)(&Bp[48 + mh][kh * 16]);
      fb0.u[0] = q0[0]; fb0.u[1] = q0[1];
      fb1.u[0] = q1[0]; fb1.u[1] = q1[1];
      fb2.u[0] = q2[0]; fb2.u[1] = q2[1];
      fb3.u[0] = q3[0]; fb3.u[1] = q3[1];
    }

    acc0 = __builtin_amdgcn_wmma_f32_16x16x32_bf16(false, af, false, fb0.v, (short)0, acc0, false, false);
    acc1 = __builtin_amdgcn_wmma_f32_16x16x32_bf16(false, af, false, fb1.v, (short)0, acc1, false, false);
    acc2 = __builtin_amdgcn_wmma_f32_16x16x32_bf16(false, af, false, fb2.v, (short)0, acc2, false, false);
    acc3 = __builtin_amdgcn_wmma_f32_16x16x32_bf16(false, af, false, fb3.v, (short)0, acc3, false, false);

    __syncthreads();
  }

  // C/D layout: lanes 0-15 -> M=j, lanes 16-31 -> M=j+8; N = lane&15
#pragma unroll
  for (int j = 0; j < 8; ++j) {
    int m = mBase + kh * 8 + j;
    float* crow = C + (long long)m * 64 + mh;
    crow[ 0] = acc0[j];
    crow[16] = acc1[j];
    crow[32] = acc2[j];
    crow[48] = acc3[j];
  }
}

// ---------------- tag finish: Klein avg -> k2p -> p2l -> logmap0 ----------------
__global__ void k_tagfinish(const float* __restrict__ Cit, float* __restrict__ xt, int nitems) {
  int r = blockIdx.x * blockDim.x + threadIdx.x;
  if (r >= nitems) return;
  const float* c = Cit + (long long)r * 64;
  float den = fmaxf(c[63], EPSF);
  float inv = 1.f / den;
  float sc2 = 0.f;
#pragma unroll
  for (int j = 0; j < 63; ++j) { float v = c[j]; sc2 += v * v; }
  float ny = sc2 * inv * inv;
  float kden = 1.f + sqrtf(fmaxf(1.f - ny, EPSF));
  float pf = inv / kden;
  float np = sc2 * pf * pf;
  float dl = 1.f - np + 1e-6f;
  float x0 = (1.f + np) / dl;
  float rf = 2.f * pf / dl;
  float rn = sqrtf(fmaxf(sc2 * rf * rf, EPSF));
  float alpha = fmaxf(x0, 1.f + 1e-7f);
  float lsc = acoshf(alpha) / rn;
  float* o = xt + (long long)r * 64;
  o[0] = 0.f;
#pragma unroll
  for (int j = 0; j < 63; ++j) o[j + 1] = lsc * rf * c[j];
}

// ---------------- pair distances: one wave32 per pair ----------------
__global__ void k_dist(const float* __restrict__ h_in, const float* __restrict__ h_gr,
                       const int* __restrict__ idx, float* __restrict__ out, int B) {
  int gt = blockIdx.x * blockDim.x + threadIdx.x;
  int pair = gt >> 5;
  int lane = gt & 31;
  if (pair >= B) return;
  int u = idx[2 * pair], v = idx[2 * pair + 1];

  {
    const float* xu = h_in + (long long)u * 64;
    const float* xv = h_in + (long long)v * 64;
    float t0 = xu[lane] * xv[lane];
    if (lane == 0) t0 = -t0;
    float part = t0 + xu[lane + 32] * xv[lane + 32];
#pragma unroll
    for (int off = 16; off; off >>= 1) part += __shfl_xor(part, off, 32);
    float d = fmaxf(-part, 1.f + 1e-7f);
    float a = acoshf(d);
    if (lane == 0) out[pair] = fminf(a * a, 50.f);
  }
  {
    const float* xu = h_gr + (long long)u * 64;
    const float* xv = h_gr + (long long)v * 64;
    float t0 = xu[lane] * xv[lane];
    if (lane == 0) t0 = -t0;
    float part = t0 + xu[lane + 32] * xv[lane + 32];
#pragma unroll
    for (int off = 16; off; off >>= 1) part += __shfl_xor(part, off, 32);
    float d = fmaxf(-part, 1.f + 1e-7f);
    float a = acoshf(d);
    if (lane == 0) out[pair] += fminf(a * a, 15.f);
  }
}

// ---------------- host orchestration ----------------

extern "C" void kernel_launch(void* const* d_in, const int* in_sizes, int n_in,
                              void* d_out, int out_size, void* d_ws, size_t ws_size,
                              hipStream_t stream) {
  const float* emb  = (const float*)d_in[0];
  const float* Tw   = (const float*)d_in[1];
  const float* ugr  = (const float*)d_in[2];
  const float* sps  = (const float*)d_in[3];
  const float* ew   = (const float*)d_in[4];
  const int*   esrc = (const int*)d_in[5];
  const int*   edst = (const int*)d_in[6];
  const int*   idx  = (const int*)d_in[7];
  float* out = (float*)d_out;

  const int D = 64;
  const long long N  = in_sizes[0] / D;          // 49152
  const int TAGS     = in_sizes[1] / D;          // 1024
  const int NU       = in_sizes[2] / D;          // 8192
  const int NI       = in_sizes[3] / TAGS;       // 40960
  const long long E  = in_sizes[4];              // 1572864
  const int B        = in_sizes[7] / 2;          // 131072

  const long long N64 = N * 64;
  float* ws   = (float*)d_ws;
  float* xt   = ws;
  float* nxt  = xt  + N64;
  float* acc  = nxt + N64;
  float* h_in = acc + N64;
  float* h_gr = h_in + N64;
  float* M    = h_gr + N64;
  float* Cit  = M + (long long)TAGS * 64;

  dim3 blk(256);
  auto gs = [](long long n) { return dim3((unsigned)((n + 255) / 256)); };

  // ---- encoder 1: structural embeddings ----
  k_zero<<<gs(N64), blk, 0, stream>>>(acc, N64);
  k_projx_logmap<<<gs(N), blk, 0, stream>>>(emb, xt, (int)N);
  {
    float* cur = xt; float* alt = nxt;
    for (int l = 0; l < 3; ++l) {
      k_zero<<<gs(N64), blk, 0, stream>>>(alt, N64);
      k_scatter<<<gs(E * 64), blk, 0, stream>>>(cur, alt, esrc, edst, ew, E);
      k_accum<<<gs(N64), blk, 0, stream>>>(acc, alt, N64);
      float* t = cur; cur = alt; alt = t;
    }
  }
  k_expmap<<<gs(N), blk, 0, stream>>>(acc, h_in, (int)N);

  // ---- tag aggregation: prep + WMMA GEMM ----
  k_tagprep<<<gs(TAGS), blk, 0, stream>>>(Tw, M, TAGS);
  k_gemm_tags<<<dim3((unsigned)(NI / 128)), blk, 0, stream>>>(sps, M, Cit, TAGS);

  // ---- encoder 2: tag-group embeddings ----
  k_zero<<<gs(N64), blk, 0, stream>>>(acc, N64);
  k_projx_logmap<<<gs(NU), blk, 0, stream>>>(ugr, xt, NU);
  k_tagfinish<<<gs(NI), blk, 0, stream>>>(Cit, xt + (long long)NU * 64, NI);
  {
    float* cur = xt; float* alt = nxt;
    for (int l = 0; l < 3; ++l) {
      k_zero<<<gs(N64), blk, 0, stream>>>(alt, N64);
      k_scatter<<<gs(E * 64), blk, 0, stream>>>(cur, alt, esrc, edst, ew, E);
      k_accum<<<gs(N64), blk, 0, stream>>>(acc, alt, N64);
      float* t = cur; cur = alt; alt = t;
    }
  }
  k_expmap<<<gs(N), blk, 0, stream>>>(acc, h_gr, (int)N);

  // ---- pair distances ----
  k_dist<<<gs((long long)B * 32), blk, 0, stream>>>(h_in, h_gr, idx, out, B);
}
